// CustomMultiheadAttention_26603027432212
// MI455X (gfx1250) — compile-verified, matrix operand order checked by reading, not run on hardware
//
#include <hip/hip_runtime.h>

// ---------------------------------------------------------------------------
// Multi-head attention w/ 2D relative-position bias for MI455X (gfx1250).
// bf16 WMMA (v_wmma_f32_16x16x32_bf16) everywhere, f32 accumulate.
// Fused scores+bias+softmax+PV kernel keeps S row-bands in registers/LDS,
// writing attn_weights exactly once (the mandatory 256 MB output).
// B=4, N=1024, D=1024, H=16, hd=64.
// ---------------------------------------------------------------------------

#define NHEAD   16
#define NTOK    1024
#define DEMB    1024
#define HD      64
#define BATCH   4

typedef __attribute__((ext_vector_type(16))) __bf16 v16bf;
typedef __attribute__((ext_vector_type(8)))  __bf16 v8bf;
typedef __attribute__((ext_vector_type(8)))  float  v8f;
typedef int v4i_ __attribute__((vector_size(16)));

// gfx1250 async global->LDS staging (guarded; sync fallback otherwise)
#if __has_builtin(__builtin_amdgcn_global_load_async_to_lds_b128) && \
    __has_builtin(__builtin_amdgcn_s_wait_asynccnt)
#define USE_ASYNC_LDS 1
#else
#define USE_ASYNC_LDS 0
#endif

// ---- WMMA fragment helpers -------------------------------------------------
// CDNA5 16-bit A-frag: lanes 0-15 -> rows M=0..15, K = {0..7, 16..23};
// lanes 16-31 -> same rows, K = {8..15, 24..31}.  B-frag mirrors with lane->N.
// Both load from row-major [rows x K] arrays (NT GEMM: C = A * B^T).

__device__ __forceinline__ v16bf frag_f32(const float* __restrict__ base,
                                          int ld, int row, int kb) {
  const float* p = base + (size_t)row * ld + kb;
  v16bf f;
#pragma unroll
  for (int i = 0; i < 8; ++i) f[i] = (__bf16)p[i];
#pragma unroll
  for (int i = 0; i < 8; ++i) f[8 + i] = (__bf16)p[16 + i];
  return f;
}

__device__ __forceinline__ v16bf frag_bf16(const __bf16* base, int ld,
                                           int row, int kb) {
  const __bf16* p = base + (size_t)row * ld + kb;
  v8bf lo = *(const v8bf*)(p);
  v8bf hi = *(const v8bf*)(p + 16);
  v16bf f;
#pragma unroll
  for (int i = 0; i < 8; ++i) { f[i] = lo[i]; f[8 + i] = hi[i]; }
  return f;
}

__device__ __forceinline__ v8f wmma_bf16(v16bf a, v16bf b, v8f c) {
  return __builtin_amdgcn_wmma_f32_16x16x32_bf16(
      /*neg_a=*/false, a, /*neg_b=*/false, b,
      /*c_mod=*/(short)0, c, /*reuse_a=*/false, /*reuse_b=*/false);
}

// ---------------------------------------------------------------------------
// Kernel 1: qkv = query @ Win^T + bin; scatter into q(bf16, pre-scaled),
// k(bf16) as [B,H,N,hd], and v transposed [B,H,hd,N] (for NT P@V GEMM).
// ---------------------------------------------------------------------------
__global__ void __launch_bounds__(256)
k_qkv(const float* __restrict__ query, const float* __restrict__ win,
      const float* __restrict__ bin,
      __bf16* __restrict__ qbf, __bf16* __restrict__ kbf,
      __bf16* __restrict__ vT) {
  const int lane = threadIdx.x & 31;
  const int wave = threadIdx.x >> 5;
  const int tile = blockIdx.x * 8 + wave;   // 12288 tiles of 32x32
  const int TN   = 3072 / 32;               // 96
  const int tm   = tile / TN;
  const int tn   = tile % TN;
  const int rowA = tm * 32;                 // [0,4096)
  const int rowB = tn * 32;                 // output col base [0,3072)
  const int lrow = lane & 15;
  const int kgrp = (lane >> 4) << 3;

  v8f acc[2][2] = {};
  for (int kb = 0; kb < DEMB; kb += 32) {
    v16bf a0 = frag_f32(query, DEMB, rowA + lrow,      kb + kgrp);
    v16bf a1 = frag_f32(query, DEMB, rowA + 16 + lrow, kb + kgrp);
    v16bf b0 = frag_f32(win,   DEMB, rowB + lrow,      kb + kgrp);
    v16bf b1 = frag_f32(win,   DEMB, rowB + 16 + lrow, kb + kgrp);
    acc[0][0] = wmma_bf16(a0, b0, acc[0][0]);
    acc[0][1] = wmma_bf16(a0, b1, acc[0][1]);
    acc[1][0] = wmma_bf16(a1, b0, acc[1][0]);
    acc[1][1] = wmma_bf16(a1, b1, acc[1][1]);
  }

  const int nIn  = lane & 15;
  const int mOff = (lane >> 4) * 8;
#pragma unroll
  for (int mi = 0; mi < 2; ++mi)
#pragma unroll
    for (int ni = 0; ni < 2; ++ni) {
      const int col  = rowB + ni * 16 + nIn;
      const float bc = bin[col];
#pragma unroll
      for (int i = 0; i < 8; ++i) {
        const int grow = rowA + mi * 16 + mOff + i;   // [0,4096)
        const int b = grow >> 10, n = grow & 1023;
        const float v = acc[mi][ni][i] + bc;
        if (col < 1024) {            // q, fold in scaling hd^-0.5 = 0.125
          const int h = col >> 6, d = col & 63;
          qbf[(((size_t)(b * NHEAD + h)) * NTOK + n) * HD + d] =
              (__bf16)(v * 0.125f);
        } else if (col < 2048) {     // k
          const int c = col - 1024, h = c >> 6, d = c & 63;
          kbf[(((size_t)(b * NHEAD + h)) * NTOK + n) * HD + d] = (__bf16)v;
        } else {                     // v, transposed [B,H,hd,N]
          const int c = col - 2048, h = c >> 6, d = c & 63;
          vT[(((size_t)(b * NHEAD + h)) * HD + d) * NTOK + n] = (__bf16)v;
        }
      }
    }
}

// ---------------------------------------------------------------------------
// Kernel 2 (fused): for one (b,h) and a 32-row band:
//   S = q@k^T + rel_bias  (full 1024 width in registers across 8 waves)
//   block-wide softmax (shuffle + LDS reductions)
//   write P (f32) once to d_out, stash bf16 copy in LDS
//   out_band = P @ v from LDS  (one 16x16 tile per wave, K=1024)
// ---------------------------------------------------------------------------
__global__ void __launch_bounds__(256)
k_attn(const __bf16* __restrict__ qbf, const __bf16* __restrict__ kbf,
       const __bf16* __restrict__ vT, const float* __restrict__ relTable,
       float* __restrict__ S, float* __restrict__ attnOut) {
  __shared__ __bf16 qs[32 * HD];        // 4 KB   q row band
  __shared__ float  tbl[64 * NHEAD];    // 4 KB   rel bias table
  __shared__ int    lut[63];            //        bucket LUT, rel in [-31,31]
  __shared__ float  redA[32 * 8];       // 1 KB   row-max partials
  __shared__ float  redB[32 * 8];       // 1 KB   row-sum partials
  __shared__ __bf16 Ps[32 * NTOK];      // 64 KB  normalized P (bf16)

  const int bh   = blockIdx.x >> 5;     // [0,64)
  const int rblk = blockIdx.x & 31;     // [0,32)
  const int rowA = rblk * 32;
  const int b = bh >> 4, h = bh & 15;
  const int lane = threadIdx.x & 31;
  const int wave = threadIdx.x >> 5;
  const int lrow = lane & 15;
  const int kgrp = (lane >> 4) << 3;
  const int nIn  = lane & 15;
  const int mOff = (lane >> 4) * 8;

  const __bf16* qg = qbf + ((size_t)bh * NTOK + rowA) * HD;
  const __bf16* kg = kbf + (size_t)bh * NTOK * HD;
  const __bf16* vg = vT  + (size_t)bh * HD * NTOK;
  float* Sp = S + (size_t)bh * NTOK * NTOK;

  // ---- stage q band (32x64 bf16 = 4KB) into LDS, async on gfx1250 --------
#if USE_ASYNC_LDS
  {
    const __bf16* src = qg + threadIdx.x * 8;        // 16B per thread
    __bf16* dst = &qs[threadIdx.x * 8];
    __builtin_amdgcn_global_load_async_to_lds_b128(
        (__attribute__((address_space(1))) v4i_*)(uintptr_t)src,
        (__attribute__((address_space(3))) v4i_*)(uint32_t)(uintptr_t)dst,
        0, 0);
  }
#else
  *(v8bf*)&qs[threadIdx.x * 8] = *(const v8bf*)(qg + threadIdx.x * 8);
#endif

  for (int i = threadIdx.x; i < 64 * NHEAD; i += 256) tbl[i] = relTable[i];
  if (threadIdx.x < 63) {
    const int d = (int)threadIdx.x - 31;
    const int a = d < 0 ? -d : d;
    int bkt;
    if (a < 16) bkt = a;
    else bkt = 16 + (int)(__logf((float)a * (1.0f / 16.0f)) *
                          (16.0f / 2.0794415416798357f));   // 16/ln(8)
    if (bkt > 31) bkt = 31;
    lut[threadIdx.x] = (d < 0 ? 32 : 0) + bkt;
  }
#if USE_ASYNC_LDS
  __builtin_amdgcn_s_wait_asynccnt(0);
#endif
  __syncthreads();

  // ---- S band: each wave covers 128 columns (2 M-tiles x 8 N-tiles) ------
  const int colBase = wave * 128;
  v8f acc[2][8] = {};
#pragma unroll
  for (int kb = 0; kb < HD; kb += 32) {
    v16bf a0 = frag_bf16(qs, HD, lrow,      kb + kgrp);
    v16bf a1 = frag_bf16(qs, HD, 16 + lrow, kb + kgrp);
#pragma unroll
    for (int ni = 0; ni < 8; ++ni) {
      v16bf bf = frag_bf16(kg, HD, colBase + ni * 16 + lrow, kb + kgrp);
      acc[0][ni] = wmma_bf16(a0, bf, acc[0][ni]);
      acc[1][ni] = wmma_bf16(a1, bf, acc[1][ni]);
    }
  }

  // ---- add relative-position bias ----------------------------------------
#pragma unroll
  for (int ni = 0; ni < 8; ++ni) {
    const int n  = colBase + ni * 16 + nIn;
    const int rn = n >> 5, cn = n & 31;
#pragma unroll
    for (int mi = 0; mi < 2; ++mi)
#pragma unroll
      for (int i = 0; i < 8; ++i) {
        const int m  = rowA + mi * 16 + mOff + i;
        const int rb = lut[(m >> 5) - rn + 31];
        const int cb = lut[(m & 31) - cn + 31];
        acc[mi][ni][i] += tbl[rb * NHEAD + h] + tbl[cb * NHEAD + h];
      }
  }

  // ---- row max (per lane -> half-wave shuffle -> LDS across waves) -------
  float rloc[16];
#pragma unroll
  for (int mi = 0; mi < 2; ++mi)
#pragma unroll
    for (int i = 0; i < 8; ++i) {
      float v = acc[mi][0][i];
#pragma unroll
      for (int ni = 1; ni < 8; ++ni) v = fmaxf(v, acc[mi][ni][i]);
#pragma unroll
      for (int off = 8; off; off >>= 1) v = fmaxf(v, __shfl_xor(v, off, 32));
      rloc[mi * 8 + i] = v;
    }
  if ((lane & 15) == 0) {
#pragma unroll
    for (int mi = 0; mi < 2; ++mi)
#pragma unroll
      for (int i = 0; i < 8; ++i)
        redA[(mi * 16 + mOff + i) * 8 + wave] = rloc[mi * 8 + i];
  }
  __syncthreads();
  float rowmax[16];
#pragma unroll
  for (int mi = 0; mi < 2; ++mi)
#pragma unroll
    for (int i = 0; i < 8; ++i) {
      const int r = mi * 16 + mOff + i;
      float v = redA[r * 8];
#pragma unroll
      for (int j = 1; j < 8; ++j) v = fmaxf(v, redA[r * 8 + j]);
      rowmax[mi * 8 + i] = v;
    }

  // ---- exp + row sum -----------------------------------------------------
#pragma unroll
  for (int mi = 0; mi < 2; ++mi)
#pragma unroll
    for (int i = 0; i < 8; ++i) {
      float s = 0.f;
#pragma unroll
      for (int ni = 0; ni < 8; ++ni) {
        const float e = __expf(acc[mi][ni][i] - rowmax[mi * 8 + i]);
        acc[mi][ni][i] = e;
        s += e;
      }
#pragma unroll
      for (int off = 8; off; off >>= 1) s += __shfl_xor(s, off, 32);
      rloc[mi * 8 + i] = s;
    }
  if ((lane & 15) == 0) {
#pragma unroll
    for (int mi = 0; mi < 2; ++mi)
#pragma unroll
      for (int i = 0; i < 8; ++i)
        redB[(mi * 16 + mOff + i) * 8 + wave] = rloc[mi * 8 + i];
  }
  __syncthreads();
  float rinv[16];
#pragma unroll
  for (int mi = 0; mi < 2; ++mi)
#pragma unroll
    for (int i = 0; i < 8; ++i) {
      const int r = mi * 16 + mOff + i;
      float v = redB[r * 8];
#pragma unroll
      for (int j = 1; j < 8; ++j) v += redB[r * 8 + j];
      rinv[mi * 8 + i] = 1.0f / v;
    }

  // ---- normalize: write attn_weights (f32, once) + bf16 copy to LDS -----
#pragma unroll
  for (int mi = 0; mi < 2; ++mi)
#pragma unroll
    for (int ni = 0; ni < 8; ++ni)
#pragma unroll
      for (int i = 0; i < 8; ++i) {
        const int lm = mi * 16 + mOff + i;
        const int n  = colBase + ni * 16 + nIn;
        const float p = acc[mi][ni][i] * rinv[mi * 8 + i];
        Sp[(size_t)(rowA + lm) * NTOK + n] = p;
        Ps[lm * NTOK + n] = (__bf16)p;
      }
  __syncthreads();

  // ---- PV: each wave owns one 16x16 output tile, K = 1024 ----------------
  const int mi2 = wave >> 2;   // 0..1 (row tile)
  const int nt  = wave & 3;    // 0..3 (col tile within hd)
  v8f o0 = {}, o1 = {};
  for (int kb = 0; kb < NTOK; kb += 64) {
    v16bf a0 = frag_bf16(Ps, NTOK, mi2 * 16 + lrow, kb + kgrp);
    v16bf b0 = frag_bf16(vg, NTOK, nt * 16 + lrow,  kb + kgrp);
    o0 = wmma_bf16(a0, b0, o0);
    v16bf a1 = frag_bf16(Ps, NTOK, mi2 * 16 + lrow, kb + 32 + kgrp);
    v16bf b1 = frag_bf16(vg, NTOK, nt * 16 + lrow,  kb + 32 + kgrp);
    o1 = wmma_bf16(a1, b1, o1);
  }
#pragma unroll
  for (int i = 0; i < 8; ++i) {
    const int m   = rowA + mi2 * 16 + mOff + i;
    const int col = h * HD + nt * 16 + nIn;
    attnOut[((size_t)(b * NTOK + m)) * DEMB + col] = o0[i] + o1[i];
  }
}

// ---------------------------------------------------------------------------
// Kernel 3: out = attnOut @ Wout^T + bout
// ---------------------------------------------------------------------------
__global__ void __launch_bounds__(256)
k_outproj(const float* __restrict__ attnOut, const float* __restrict__ wout,
          const float* __restrict__ bout, float* __restrict__ out) {
  const int lane = threadIdx.x & 31;
  const int wave = threadIdx.x >> 5;
  const int tile = blockIdx.x * 8 + wave;   // 4096 tiles (128 M x 32 N)
  const int tm = tile >> 5, tn = tile & 31;
  const int rowA = tm * 32;                 // [0,4096)
  const int rowB = tn * 32;                 // [0,1024)
  const int lrow = lane & 15, kgrp = (lane >> 4) << 3;

  v8f acc[2][2] = {};
  for (int kb = 0; kb < DEMB; kb += 32) {
    v16bf a0 = frag_f32(attnOut, DEMB, rowA + lrow,      kb + kgrp);
    v16bf a1 = frag_f32(attnOut, DEMB, rowA + 16 + lrow, kb + kgrp);
    v16bf b0 = frag_f32(wout,    DEMB, rowB + lrow,      kb + kgrp);
    v16bf b1 = frag_f32(wout,    DEMB, rowB + 16 + lrow, kb + kgrp);
    acc[0][0] = wmma_bf16(a0, b0, acc[0][0]);
    acc[0][1] = wmma_bf16(a0, b1, acc[0][1]);
    acc[1][0] = wmma_bf16(a1, b0, acc[1][0]);
    acc[1][1] = wmma_bf16(a1, b1, acc[1][1]);
  }

  const int nIn  = lane & 15;
  const int mOff = (lane >> 4) * 8;
#pragma unroll
  for (int mi = 0; mi < 2; ++mi)
#pragma unroll
    for (int ni = 0; ni < 2; ++ni) {
      const int col = rowB + ni * 16 + nIn;
      const float bc = bout[col];
#pragma unroll
      for (int i = 0; i < 8; ++i) {
        const int m = rowA + mi * 16 + mOff + i;
        out[(size_t)m * DEMB + col] = acc[mi][ni][i] + bc;
      }
    }
}

// ---------------------------------------------------------------------------
extern "C" void kernel_launch(void* const* d_in, const int* in_sizes, int n_in,
                              void* d_out, int out_size, void* d_ws,
                              size_t ws_size, hipStream_t stream) {
  (void)in_sizes; (void)n_in; (void)out_size; (void)ws_size;
  const float* query = (const float*)d_in[0];
  // d_in[1] (key) and d_in[2] (value) are ignored by the reference.
  const float* win   = (const float*)d_in[3];
  const float* bin   = (const float*)d_in[4];
  const float* wout  = (const float*)d_in[5];
  const float* bout  = (const float*)d_in[6];
  const float* relT  = (const float*)d_in[7];

  const size_t QKV = (size_t)BATCH * NHEAD * NTOK * HD;   // 4 Mi elems
  __bf16* qbf = (__bf16*)d_ws;
  __bf16* kbf = qbf + QKV;
  __bf16* vT  = kbf + QKV;
  float* attnOut = (float*)(vT + QKV);                    // 16 MB f32

  float* out = (float*)d_out;
  float* S   = out + (size_t)BATCH * NTOK * DEMB;         // attn_weights

  k_qkv    <<<1536, 256, 0, stream>>>(query, win, bin, qbf, kbf, vT);
  k_attn   <<<2048, 256, 0, stream>>>(qbf, kbf, vT, relT, S, attnOut);
  k_outproj<<<512,  256, 0, stream>>>(attnOut, wout, bout, out);
}
